// VarianceAdaptor_13580686590627
// MI455X (gfx1250) — matrix operand order repaired and checked by reading, not compiled, not run on hardware
//
#include <hip/hip_runtime.h>
#include <hip/hip_bf16.h>

typedef _Float16 h16;
typedef __attribute__((ext_vector_type(16))) _Float16 v16h;
typedef __attribute__((ext_vector_type(8)))  _Float16 v8h;
typedef __attribute__((ext_vector_type(8)))  float    v8f;

#define BATCH 16
#define SEQ   512
#define DM    256
#define MAXT  2048

// ---------------------------------------------------------------------------
// Duration cumsum + frame->token index scatter (length regulator indices).
// One block per batch element, 512 threads; Hillis-Steele inclusive scan.
// idx[b][t] = token s with cum[s-1] <= t < cum[s], else -1 (masked frame).
// ---------------------------------------------------------------------------
__global__ __launch_bounds__(SEQ)
void k_duridx(const int* __restrict__ D, int* __restrict__ cum, int* __restrict__ idx) {
    int b = blockIdx.x, s = threadIdx.x;
    __shared__ int sc[SEQ];
    int d = D[b * SEQ + s];
    sc[s] = d;
    __syncthreads();
    for (int off = 1; off < SEQ; off <<= 1) {
        int v = (s >= off) ? sc[s - off] : 0;
        __syncthreads();
        sc[s] += v;
        __syncthreads();
    }
    int c = sc[s];
    cum[b * SEQ + s] = c;
    for (int t = s; t < MAXT; t += SEQ) idx[b * MAXT + t] = -1;
    __syncthreads();
    for (int t = c - d; t < c; ++t) idx[b * MAXT + t] = s;   // d in [0,4)
}

// ---------------------------------------------------------------------------
// Shuffle conv weights W[tap][ci][co] (f32, WIO) into per-lane-contiguous f16
// B-fragments matching the CDNA5 16-bit B operand layout:
//   lane group g=lane/16 holds K = g*8 + i (i < 8), 16 + g*8 + (i-8) (i >= 8), col n = lane%16.
// Fragment id = (((tap*8+kt)*16 + ntile)*32 + lane)*16 + i, per conv (6 convs).
// ---------------------------------------------------------------------------
__global__ __launch_bounds__(256)
void k_wfrag(const float* __restrict__ w0, const float* __restrict__ w1,
             const float* __restrict__ w2, const float* __restrict__ w3,
             const float* __restrict__ w4, const float* __restrict__ w5,
             h16* __restrict__ wfrag) {
    int id   = blockIdx.x * 256 + threadIdx.x;      // < 6 * 196608
    int conv = id / 196608;
    int rem  = id - conv * 196608;
    int i     = rem & 15;
    int lane  = (rem >> 4) & 31;
    int ntile = (rem >> 9) & 15;
    int kt    = (rem >> 13) & 7;
    int tap   = rem >> 16;
    int g = lane >> 4, n = lane & 15;
    const float* W = (conv == 0) ? w0 : (conv == 1) ? w1 : (conv == 2) ? w2
                   : (conv == 3) ? w3 : (conv == 4) ? w4 : w5;
    int k = kt * 32 + ((i < 8) ? (g * 8 + i) : (16 + g * 8 + (i - 8)));
    wfrag[id] = (h16)W[tap * (DM * DM) + k * DM + ntile * 16 + n];
}

// ---------------------------------------------------------------------------
// Zero one (padded) row of a f16 activation buffer for every batch.
// Guard rows make conv boundary loads unconditionally valid (SAME padding).
// ---------------------------------------------------------------------------
__global__ __launch_bounds__(DM)
void k_zerorow(h16* __restrict__ base, int strideRows, int row) {
    base[((size_t)blockIdx.x * strideRows + row) * DM + threadIdx.x] = (h16)0.f;
}

// ---------------------------------------------------------------------------
// f32 -> f16 cast into padded [B, SEQ+2, DM] buffer (data at row t+1).
// ---------------------------------------------------------------------------
__global__ __launch_bounds__(256)
void k_cast16(const float* __restrict__ x, h16* __restrict__ y) {
    int i = blockIdx.x * 256 + threadIdx.x;         // < B*SEQ*DM
    int c = i & 255;
    int t = (i >> 8) & (SEQ - 1);
    int b = i >> 17;
    y[((size_t)b * (SEQ + 2) + t + 1) * DM + c] = (h16)x[i];
}

// ---------------------------------------------------------------------------
// Length-regulator gather into padded [B, MAXT+2, DM] f16 buffer (row t+1).
// ---------------------------------------------------------------------------
__global__ __launch_bounds__(256)
void k_gather16(const float* __restrict__ H, const int* __restrict__ idx,
                h16* __restrict__ he) {
    int i = blockIdx.x * 256 + threadIdx.x;         // < B*MAXT*DM
    int c = i & 255;
    int bt = i >> 8;                                // b*MAXT + t
    int t = bt & (MAXT - 1);
    int b = bt >> 11;
    int s = idx[bt];
    float v = (s >= 0) ? H[((size_t)(b * SEQ + s)) * DM + c] : 0.f;
    he[((size_t)b * (MAXT + 2) + t + 1) * DM + c] = (h16)v;
}

// ---------------------------------------------------------------------------
// CDNA5 16-bit A operand fragment load (16x32, row m = lane%16):
//   lane group g holds K-locals {g*8..g*8+7} U {16+g*8..16+g*8+7}
//   -> two contiguous 16-byte loads from the padded [T+2,256] f16 row.
// ---------------------------------------------------------------------------
__device__ __forceinline__ v16h load_a(const h16* __restrict__ xrow, int kbase, int g) {
    v8h lo = *(const v8h*)(xrow + kbase + g * 8);
    v8h hi = *(const v8h*)(xrow + kbase + 16 + g * 8);
    return __builtin_shufflevector(lo, hi, 0,1,2,3,4,5,6,7,8,9,10,11,12,13,14,15);
}

// ---------------------------------------------------------------------------
// conv1d(k=3, 256->256, SAME) + bias + ReLU, padded f16 in / padded f16 out.
// Block = 16 time steps x 256 channels; 8 waves, 2 N-tiles each;
// K-loop = 3 taps x 8 K-tiles of 32 -> 48 v_wmma_f32_16x16x32_f16 per wave.
// Guard rows (zeros) make all A loads unconditional -> no EXEC divergence.
// ---------------------------------------------------------------------------
__global__ __launch_bounds__(256)
void k_conv_relu(const h16* __restrict__ x, const h16* __restrict__ wf,
                 const float* __restrict__ bias, h16* __restrict__ y, int T) {
    const int b    = blockIdx.y;
    const int t0   = blockIdx.x << 4;
    const int lane = threadIdx.x & 31;
    const int wave = threadIdx.x >> 5;
    const int g    = lane >> 4, ln = lane & 15;
    const int nt0  = wave * 2, nt1 = nt0 + 1;
    const h16* xb  = x + (size_t)b * (T + 2) * DM;
    v8f a0 = {}, a1 = {};
    for (int tap = 0; tap < 3; ++tap) {
        const h16* xrow = xb + (size_t)(t0 + ln + tap) * DM;   // (t0+ln-1+tap)+1 pad
        for (int kt = 0; kt < 8; ++kt) {
            v16h a = load_a(xrow, kt * 32, g);
            const h16* wb = wf + (size_t)(tap * 8 + kt) * 8192;
            v16h b0 = *(const v16h*)(wb + (size_t)(nt0 * 32 + lane) * 16);
            v16h b1 = *(const v16h*)(wb + (size_t)(nt1 * 32 + lane) * 16);
            a0 = __builtin_amdgcn_wmma_f32_16x16x32_f16(false, a, false, b0, (short)0, a0, false, false);
            a1 = __builtin_amdgcn_wmma_f32_16x16x32_f16(false, a, false, b1, (short)0, a1, false, false);
        }
    }
    float bi0 = bias[nt0 * 16 + ln], bi1 = bias[nt1 * 16 + ln];
    h16* yb = y + ((size_t)b * (T + 2) + t0 + 1) * DM;
    for (int r = 0; r < 8; ++r) {
        int m = r + 8 * g;                          // C/D layout: m = vgpr + 8*(lane/16)
        float v0 = a0[r] + bi0; v0 = v0 > 0.f ? v0 : 0.f;
        float v1 = a1[r] + bi1; v1 = v1 > 0.f ? v1 : 0.f;
        yb[(size_t)m * DM + nt0 * 16 + ln] = (h16)v0;
        yb[(size_t)m * DM + nt1 * 16 + ln] = (h16)v1;
    }
}

// ---------------------------------------------------------------------------
// conv1d(k=3) + bias + ReLU fused with Linear(256->1): pred[b,t].
// Same GEMM core; epilogue dots with Wl, shfl-reduces across the 16 N-lanes
// (xor 1/2/4/8 stays within each 16-lane half = one N-tile), then LDS-adds
// across waves/N-tiles. Input x is padded with stride strideRows = padT+2.
// ---------------------------------------------------------------------------
__global__ __launch_bounds__(256)
void k_conv_pred(const h16* __restrict__ x, const h16* __restrict__ wf,
                 const float* __restrict__ bias, const float* __restrict__ Wl,
                 const float* __restrict__ bl, float* __restrict__ pred,
                 int T, int strideRows) {
    __shared__ float sred[16];
    const int b    = blockIdx.y;
    const int t0   = blockIdx.x << 4;
    const int lane = threadIdx.x & 31;
    const int wave = threadIdx.x >> 5;
    const int g    = lane >> 4, ln = lane & 15;
    const int nt0  = wave * 2, nt1 = nt0 + 1;
    if (threadIdx.x < 16) sred[threadIdx.x] = 0.f;
    __syncthreads();
    const h16* xb = x + (size_t)b * strideRows * DM;
    v8f a0 = {}, a1 = {};
    for (int tap = 0; tap < 3; ++tap) {
        const h16* xrow = xb + (size_t)(t0 + ln + tap) * DM;
        for (int kt = 0; kt < 8; ++kt) {
            v16h a = load_a(xrow, kt * 32, g);
            const h16* wb = wf + (size_t)(tap * 8 + kt) * 8192;
            v16h b0 = *(const v16h*)(wb + (size_t)(nt0 * 32 + lane) * 16);
            v16h b1 = *(const v16h*)(wb + (size_t)(nt1 * 32 + lane) * 16);
            a0 = __builtin_amdgcn_wmma_f32_16x16x32_f16(false, a, false, b0, (short)0, a0, false, false);
            a1 = __builtin_amdgcn_wmma_f32_16x16x32_f16(false, a, false, b1, (short)0, a1, false, false);
        }
    }
    float bi0 = bias[nt0 * 16 + ln], bi1 = bias[nt1 * 16 + ln];
    float wl0 = Wl[nt0 * 16 + ln],   wl1 = Wl[nt1 * 16 + ln];
    for (int r = 0; r < 8; ++r) {
        float v0 = a0[r] + bi0; v0 = v0 > 0.f ? v0 : 0.f;
        float v1 = a1[r] + bi1; v1 = v1 > 0.f ? v1 : 0.f;
        float p = v0 * wl0 + v1 * wl1;
        p += __shfl_xor(p, 1, 32);
        p += __shfl_xor(p, 2, 32);
        p += __shfl_xor(p, 4, 32);
        p += __shfl_xor(p, 8, 32);
        if (ln == 0) atomicAdd(&sred[r + 8 * g], p);
    }
    __syncthreads();
    if (threadIdx.x < 16)
        pred[(size_t)b * T + t0 + threadIdx.x] = sred[threadIdx.x] + bl[0];
}

// ---------------------------------------------------------------------------
// H_adapted = He + P_gt*Wp + bp + E_gt*We + be  (He regathered from H in f32)
// ---------------------------------------------------------------------------
__global__ __launch_bounds__(256)
void k_adapt(const float* __restrict__ H, const int* __restrict__ idx,
             const float* __restrict__ Pgt, const float* __restrict__ Egt,
             const float* __restrict__ Wp, const float* __restrict__ bp,
             const float* __restrict__ We, const float* __restrict__ be,
             float* __restrict__ out) {
    int i = blockIdx.x * 256 + threadIdx.x;         // < B*MAXT*DM
    int c = i & 255;
    int bt = i >> 8;                                // b*MAXT + t
    int s = idx[bt];
    int b = bt >> 11;
    float he = (s >= 0) ? H[((size_t)(b * SEQ + s)) * DM + c] : 0.f;
    out[i] = he + Pgt[bt] * Wp[c] + bp[c] + Egt[bt] * We[c] + be[c];
}

// ---------------------------------------------------------------------------
extern "C" void kernel_launch(void* const* d_in, const int* in_sizes, int n_in,
                              void* d_out, int out_size, void* d_ws, size_t ws_size,
                              hipStream_t stream) {
    (void)in_sizes; (void)n_in; (void)out_size; (void)ws_size;
    // setup_inputs() order: H, P_gt, E_gt, params{dur{W1,b1,W2,b2,Wl,bl}, pitch{...},
    //                       energy{...}, Wp, bp, We, be}, D_gt
    const float* H   = (const float*)d_in[0];
    const float* Pgt = (const float*)d_in[1];
    const float* Egt = (const float*)d_in[2];
    const float* dW1 = (const float*)d_in[3];
    const float* db1 = (const float*)d_in[4];
    const float* dW2 = (const float*)d_in[5];
    const float* db2 = (const float*)d_in[6];
    const float* dWl = (const float*)d_in[7];
    const float* dbl = (const float*)d_in[8];
    const float* pW1 = (const float*)d_in[9];
    const float* pb1 = (const float*)d_in[10];
    const float* pW2 = (const float*)d_in[11];
    const float* pb2 = (const float*)d_in[12];
    const float* pWl = (const float*)d_in[13];
    const float* pbl = (const float*)d_in[14];
    const float* eW1 = (const float*)d_in[15];
    const float* eb1 = (const float*)d_in[16];
    const float* eW2 = (const float*)d_in[17];
    const float* eb2 = (const float*)d_in[18];
    const float* eWl = (const float*)d_in[19];
    const float* ebl = (const float*)d_in[20];
    const float* Wp  = (const float*)d_in[21];
    const float* bp  = (const float*)d_in[22];
    const float* We  = (const float*)d_in[23];
    const float* be  = (const float*)d_in[24];
    const int*   Dgt = (const int*)d_in[25];

    // workspace layout (bytes); padded activation buffers have T+2 rows/batch
    char* ws = (char*)d_ws;
    const size_t off_cum   = 0;
    const size_t off_idx   = off_cum   + (size_t)BATCH*SEQ*4;
    const size_t off_h16   = off_idx   + (size_t)BATCH*MAXT*4;
    const size_t off_he16  = off_h16   + (size_t)BATCH*(SEQ +2)*DM*2;
    const size_t off_h1    = off_he16  + (size_t)BATCH*(MAXT+2)*DM*2;
    const size_t off_wfrag = off_h1    + (size_t)BATCH*(MAXT+2)*DM*2;
    int*  cum   = (int*)(ws + off_cum);
    int*  idx   = (int*)(ws + off_idx);
    h16*  H16   = (h16*)(ws + off_h16);
    h16*  He16  = (h16*)(ws + off_he16);
    h16*  h1    = (h16*)(ws + off_h1);
    h16*  wfrag = (h16*)(ws + off_wfrag);
    const size_t WF = 196608;  // halves per conv fragment block

    float* out  = (float*)d_out;
    float* Had  = out;                               // [16,2048,256]
    float* Dp   = Had + (size_t)BATCH*MAXT*DM;       // [16,512]
    float* Pp   = Dp  + (size_t)BATCH*SEQ;           // [16,2048]
    float* Ep   = Pp  + (size_t)BATCH*MAXT;          // [16,2048]

    // 1. weight fragment shuffle (6 convs)
    k_wfrag<<<(6 * (int)WF) / 256, 256, 0, stream>>>(dW1, dW2, pW1, pW2, eW1, eW2, wfrag);
    // 2. durations -> cumsum + frame index
    k_duridx<<<BATCH, SEQ, 0, stream>>>(Dgt, cum, idx);
    // 3. guard rows (SAME-padding zeros) for all padded activation buffers
    k_zerorow<<<BATCH, DM, 0, stream>>>(H16,  SEQ + 2,  0);
    k_zerorow<<<BATCH, DM, 0, stream>>>(H16,  SEQ + 2,  SEQ + 1);
    k_zerorow<<<BATCH, DM, 0, stream>>>(He16, MAXT + 2, 0);
    k_zerorow<<<BATCH, DM, 0, stream>>>(He16, MAXT + 2, MAXT + 1);
    k_zerorow<<<BATCH, DM, 0, stream>>>(h1,   MAXT + 2, 0);
    k_zerorow<<<BATCH, DM, 0, stream>>>(h1,   MAXT + 2, SEQ + 1);   // dur pad
    k_zerorow<<<BATCH, DM, 0, stream>>>(h1,   MAXT + 2, MAXT + 1);  // pitch/energy pad
    // 4. activation casts/gathers into padded buffers
    k_cast16<<<(BATCH*SEQ*DM) / 256, 256, 0, stream>>>(H, H16);
    k_gather16<<<(BATCH*MAXT*DM) / 256, 256, 0, stream>>>(H, idx, He16);

    // 5. duration predictor: conv1 -> h1 ; conv2 fused with Linear -> D_pred
    //    (h1 written with SEQ+2 stride here; buffer sized MAXT+2 so it serves all predictors)
    k_conv_relu<<<dim3(SEQ/16, BATCH), 256, 0, stream>>>(H16, wfrag + 0*WF, db1, h1, SEQ);
    // note: conv_relu used T to derive both read and write strides; dur wrote
    // with T=SEQ stride, so the fused head must read with SEQ+2 stride:
    k_conv_pred<<<dim3(SEQ/16, BATCH), 256, 0, stream>>>(h1,  wfrag + 1*WF, db2, dWl, dbl, Dp, SEQ, SEQ + 2);
    // 6. pitch predictor on expanded hidden
    k_conv_relu<<<dim3(MAXT/16, BATCH), 256, 0, stream>>>(He16, wfrag + 2*WF, pb1, h1, MAXT);
    k_conv_pred<<<dim3(MAXT/16, BATCH), 256, 0, stream>>>(h1,   wfrag + 3*WF, pb2, pWl, pbl, Pp, MAXT, MAXT + 2);
    // 7. energy predictor
    k_conv_relu<<<dim3(MAXT/16, BATCH), 256, 0, stream>>>(He16, wfrag + 4*WF, eb1, h1, MAXT);
    k_conv_pred<<<dim3(MAXT/16, BATCH), 256, 0, stream>>>(h1,   wfrag + 5*WF, eb2, eWl, ebl, Ep, MAXT, MAXT + 2);
    // 8. adapted hidden
    k_adapt<<<(BATCH*MAXT*DM) / 256, 256, 0, stream>>>(H, idx, Pgt, Egt, Wp, bp, We, be, Had);
}
